// BIGRU_12128987644545
// MI455X (gfx1250) — matrix operand order repaired and checked by reading, not compiled, not run on hardware
//
#include <hip/hip_runtime.h>
#include <hip/hip_bf16.h>

// ---------------------------------------------------------------------------
// BiGRU stack for MI455X (gfx1250): bf16 WMMA GEMMs + f32 gate math.
//  - gi GEMM: async-to-LDS staged 128x128 block tiles (double buffered,
//    GLOBAL_LOAD_ASYNC_TO_LDS_B128 + ASYNCcnt), 32x64 per wave
//  - recurrence: per-step kernel, each wave owns a unique w_hh column slice
// ---------------------------------------------------------------------------

typedef __attribute__((ext_vector_type(16))) __bf16 v16bf;
typedef __attribute__((ext_vector_type(8)))  __bf16 v8bf;
typedef __attribute__((ext_vector_type(8)))  float  v8f;

#define DEVI __device__ __forceinline__

constexpr int NB  = 64;        // batch
constexpr int T   = 150;       // timesteps
constexpr int CIN = 150;       // input channels
constexpr int H   = 1024;      // hidden
constexpr int G3  = 3 * H;     // 3072 gates
constexpr int NC  = 60;        // classes
constexpr int KP0 = 160;       // CIN padded to multiple of 32
constexpr int K2H = 2 * H;     // 2048
constexpr int ROWS = T * NB;   // 9600
constexpr int LPAD = 40;       // LDS row stride (32 + 8 pad) to dodge bank conflicts

// Build a v16bf A/B fragment from two contiguous 8-element (16B) runs.
DEVI v16bf frag2(const __bf16* p0, const __bf16* p1) {
  v8bf lo = *reinterpret_cast<const v8bf*>(p0);
  v8bf hi = *reinterpret_cast<const v8bf*>(p1);
  return __builtin_shufflevector(lo, hi, 0,1,2,3,4,5,6,7,8,9,10,11,12,13,14,15);
}

DEVI v8f wmma_bf(v16bf a, v16bf b, v8f c) {
  return __builtin_amdgcn_wmma_f32_16x16x32_bf16(
      /*neg_a=*/false, a, /*neg_b=*/false, b,
      /*c_mod=*/(short)0, c, /*reuse_a=*/false, /*reuse_b=*/false);
}

DEVI float sigf(float x) { return 1.0f / (1.0f + __expf(-x)); }

// CDNA5 async tensor path: per-lane 16B global -> LDS copy, tracked by ASYNCcnt.
// LDS byte address = low 32 bits of the generic shared pointer (flat-LDS
// aperture truncation, ISA sec 10.2).
DEVI void async_ld16(void* lds_ptr, const void* gaddr) {
  unsigned lds_off = (unsigned)(uintptr_t)lds_ptr;
  asm volatile("global_load_async_to_lds_b128 %0, %1, off"
               :: "v"(lds_off), "v"(gaddr)
               : "memory");
}
DEVI void wait_async0() {
  asm volatile("s_wait_asynccnt 0x0" ::: "memory");
}

// ---------------------------------------------------------------------------
// BatchNorm statistics: one block per channel, reduce over N*T = 9600.
// ---------------------------------------------------------------------------
__global__ void bn_stats_k(const float* __restrict__ x,
                           float* __restrict__ meanp, float* __restrict__ rstdp) {
  const int c = blockIdx.x, tid = threadIdx.x;
  float s = 0.f, sq = 0.f;
  for (int i = tid; i < NB * T; i += 256) {
    float v = x[(size_t)i * CIN + c];
    s += v; sq += v * v;
  }
  __shared__ float ss[256], sb[256];
  ss[tid] = s; sb[tid] = sq;
  __syncthreads();
  for (int k = 128; k > 0; k >>= 1) {
    if (tid < k) { ss[tid] += ss[tid + k]; sb[tid] += sb[tid + k]; }
    __syncthreads();
  }
  if (tid == 0) {
    float m = ss[0] / (float)(NB * T);
    float v = sb[0] / (float)(NB * T) - m * m;   // biased var (train mode)
    meanp[c] = m;
    rstdp[c] = rsqrtf(v + 1e-5f);
  }
}

// Normalize + transpose [N,T,CIN] -> bf16 [T,N,KP0] (zero-padded K).
__global__ void bn_apply_k(const float* __restrict__ x,
                           const float* __restrict__ meanp, const float* __restrict__ rstdp,
                           const float* __restrict__ gamma, const float* __restrict__ beta,
                           __bf16* __restrict__ xn) {
  int idx = blockIdx.x * 256 + threadIdx.x;
  if (idx >= T * NB * KP0) return;
  int k = idx % KP0, r = idx / KP0;
  int n = r % NB, t = r / NB;
  float v = 0.f;
  if (k < CIN)
    v = (x[((size_t)n * T + t) * CIN + k] - meanp[k]) * rstdp[k] * gamma[k] + beta[k];
  xn[idx] = (__bf16)v;
}

// ---------------------------------------------------------------------------
// Weight conversion to bf16
// ---------------------------------------------------------------------------
__global__ void cvt_k(const float* __restrict__ s, __bf16* __restrict__ d, int n) {
  int i = blockIdx.x * 256 + threadIdx.x;
  if (i < n) d[i] = (__bf16)s[i];
}

__global__ void cvt_pad_k(const float* __restrict__ s, __bf16* __restrict__ d,
                          int rows, int ks, int kd) {
  int i = blockIdx.x * 256 + threadIdx.x;
  if (i >= rows * kd) return;
  int k = i % kd, r = i / kd;
  d[i] = (k < ks) ? (__bf16)s[(size_t)r * ks + k] : (__bf16)0.f;
}

// ---------------------------------------------------------------------------
// Input-projection GEMM: gi[d, row, g] = A[row,:] . W[d,g,:] + b_ih[d,g]
// A: [ROWS, K] bf16.  W: [2, G3, K] bf16.  gi: [2, ROWS, G3] bf16.
// Block = 256 (8 waves). Block tile 128x128. K=32 panels are DMA'd into LDS
// with GLOBAL_LOAD_ASYNC_TO_LDS_B128 (double buffered, overlapped with WMMA).
// Wave tile 32x64 (2 row tiles x 4 col tiles = 8 WMMA / chunk).
// Grid: (G3/128, ROWS/128, 2)
// ---------------------------------------------------------------------------
__global__ void gi_gemm_k(const __bf16* __restrict__ A, const __bf16* __restrict__ W,
                          const float* __restrict__ bih, __bf16* __restrict__ gi, int K) {
  __shared__ __bf16 sA[2][128 * LPAD];
  __shared__ __bf16 sB[2][128 * LPAD];

  const int tid = threadIdx.x;
  const int lane = tid & 31, wv = tid >> 5;
  const int d = blockIdx.z;
  const int rowBase0 = blockIdx.y * 128;
  const int colBase0 = blockIdx.x * 128;
  const int rowG = (wv & 3) * 32;     // wave's row group within the block tile
  const int colG = (wv >> 2) * 64;    // wave's col group within the block tile
  const int l16 = lane & 15, hi = lane >> 4;

  const __bf16* Wd = W + (size_t)d * G3 * K;

  v8f acc[2][4];
#pragma unroll
  for (int rt = 0; rt < 2; rt++)
#pragma unroll
    for (int ct = 0; ct < 4; ct++) acc[rt][ct] = {};

  // Issue the async DMA of one K=32 panel pair (A: 8KB, B: 8KB) into buffer.
  // Each thread owns 2 A units + 2 B units of 16B.
  auto issue_stage = [&](int k0, int buf) {
#pragma unroll
    for (int r = 0; r < 2; r++) {
      int u = r * 256 + tid, row = u >> 2, seg = u & 3;
      async_ld16(&sA[buf][row * LPAD + seg * 8],
                 (const void*)(A  + (size_t)(rowBase0 + row) * K + k0 + seg * 8));
      async_ld16(&sB[buf][row * LPAD + seg * 8],
                 (const void*)(Wd + (size_t)(colBase0 + row) * K + k0 + seg * 8));
    }
  };

  const int nchunks = K / 32;
  issue_stage(0, 0);
  wait_async0();
  __syncthreads();

  int buf = 0;
  for (int c = 0; c < nchunks; c++) {
    if (c + 1 < nchunks) issue_stage((c + 1) * 32, buf ^ 1);

    v16bf af[2];
#pragma unroll
    for (int rt = 0; rt < 2; rt++) {
      const __bf16* p = &sA[buf][(rowG + rt * 16 + l16) * LPAD + (hi ? 16 : 0)];
      af[rt] = frag2(p, p + 8);
    }
#pragma unroll
    for (int ct = 0; ct < 4; ct++) {
      const __bf16* p = &sB[buf][(colG + ct * 16 + l16) * LPAD + (hi ? 16 : 0)];
      v16bf bf_ = frag2(p, p + 8);
#pragma unroll
      for (int rt = 0; rt < 2; rt++) acc[rt][ct] = wmma_bf(af[rt], bf_, acc[rt][ct]);
    }
    // A wave's WMMAs force its LDS reads to complete before it can reach the
    // barrier, so once all waves arrive, buf^1 writes can't race readers.
    if (c + 1 < nchunks) {
      wait_async0();
      __syncthreads();
    }
    buf ^= 1;
  }

  __bf16* outp = gi + (size_t)d * ROWS * G3;
#pragma unroll
  for (int rt = 0; rt < 2; rt++) {
    int rr = rowBase0 + rowG + rt * 16 + (hi ? 8 : 0);
#pragma unroll
    for (int ct = 0; ct < 4; ct++) {
      int col = colBase0 + colG + ct * 16 + l16;
      float bias = bih[d * G3 + col];
#pragma unroll
      for (int i = 0; i < 8; i++)
        outp[(size_t)(rr + i) * G3 + col] = (__bf16)(acc[rt][ct][i] + bias);
    }
  }
}

// ---------------------------------------------------------------------------
// One GRU timestep (both directions). Each wave owns a UNIQUE 16-column slice
// of the hidden state for all 3 gates and all 64 batch rows, so every w_hh
// fragment is read exactly once per step (w_hh is the per-step latency: 12.6MB
// bf16, L2-resident). B fragments are register-reused across 4 row tiles.
// Block = 256 (8 waves x 16 cols = 128 cols). Grid: (H/128, 1, 2)
// ---------------------------------------------------------------------------
__global__ void gru_step_k(const __bf16* __restrict__ hin_b, const float* __restrict__ hin_f,
                           __bf16* __restrict__ hout_b, float* __restrict__ hout_f,
                           const __bf16* __restrict__ whh, const float* __restrict__ bhh,
                           const __bf16* __restrict__ gi, __bf16* __restrict__ seq, int t) {
  const int lane = threadIdx.x & 31, wv = threadIdx.x >> 5;
  const int d = blockIdx.z;
  const int cBase = blockIdx.x * 128 + wv * 16;
  const int l16 = lane & 15, hi = lane >> 4;

  const __bf16* hb = hin_b + (size_t)d * NB * H;
  const __bf16* wp = whh + (size_t)d * G3 * H + (hi ? 16 : 0);

  v8f acc[3][4];
#pragma unroll
  for (int g = 0; g < 3; g++)
#pragma unroll
    for (int rt = 0; rt < 4; rt++) acc[g][rt] = {};

#pragma unroll 2
  for (int k0 = 0; k0 < H; k0 += 32) {
    v16bf af[4];
#pragma unroll
    for (int rt = 0; rt < 4; rt++) {
      const __bf16* p = hb + (size_t)(rt * 16 + l16) * H + (hi ? 16 : 0) + k0;
      af[rt] = frag2(p, p + 8);
    }
#pragma unroll
    for (int g = 0; g < 3; g++) {
      const __bf16* p = wp + (size_t)(g * H + cBase + l16) * H + k0;
      if (k0 + 64 < H) __builtin_prefetch((const void*)(p + 64), 0, 1);
      v16bf bf_ = frag2(p, p + 8);
#pragma unroll
      for (int rt = 0; rt < 4; rt++) acc[g][rt] = wmma_bf(af[rt], bf_, acc[g][rt]);
    }
  }

  const int j = cBase + l16;                 // hidden column (fixed per lane)
  const float bhr = bhh[d * G3 + j];
  const float bhz = bhh[d * G3 + H + j];
  const float bhn = bhh[d * G3 + 2 * H + j];

#pragma unroll
  for (int rt = 0; rt < 4; rt++) {
    int rr = rt * 16 + (hi ? 8 : 0);
#pragma unroll
    for (int i = 0; i < 8; i++) {
      int n = rr + i;
      const __bf16* gp = gi + (size_t)d * ROWS * G3 + (size_t)(t * NB + n) * G3;
      float gr = (float)gp[j];
      float gz = (float)gp[H + j];
      float gn = (float)gp[2 * H + j];
      float hp = hin_f[(size_t)d * NB * H + (size_t)n * H + j];
      float r  = sigf(gr + acc[0][rt][i] + bhr);
      float z  = sigf(gz + acc[1][rt][i] + bhz);
      float nn = tanhf(gn + r * (acc[2][rt][i] + bhn));
      float hn = (1.f - z) * nn + z * hp;
      hout_f[(size_t)d * NB * H + (size_t)n * H + j] = hn;
      __bf16 hbv = (__bf16)hn;
      hout_b[(size_t)d * NB * H + (size_t)n * H + j] = hbv;
      seq[(size_t)t * NB * K2H + (size_t)n * K2H + d * H + j] = hbv;
    }
  }
}

// ---------------------------------------------------------------------------
// Final FC: out[n,c] = last[n,:] . fc_w[c,:] + fc_b[c]   (64x60, tiny)
// ---------------------------------------------------------------------------
__global__ void fc_k(const __bf16* __restrict__ last, const float* __restrict__ fw,
                     const float* __restrict__ fb, float* __restrict__ out) {
  int idx = blockIdx.x * 256 + threadIdx.x;
  if (idx >= NB * NC) return;
  int c = idx % NC, n = idx / NC;
  float s = 0.f;
  for (int k = 0; k < K2H; k++) s += (float)last[(size_t)n * K2H + k] * fw[(size_t)c * K2H + k];
  out[idx] = s + fb[c];
}

// ---------------------------------------------------------------------------
// Host launcher
// ---------------------------------------------------------------------------
extern "C" void kernel_launch(void* const* d_in, const int* in_sizes, int n_in,
                              void* d_out, int out_size, void* d_ws, size_t ws_size,
                              hipStream_t stream) {
  (void)in_sizes; (void)n_in; (void)out_size; (void)ws_size;

  const float* x     = (const float*)d_in[0];
  const float* gamma = (const float*)d_in[1];
  const float* beta  = (const float*)d_in[2];
  const float* wih0  = (const float*)d_in[3];   // [2,3072,150]
  const float* whh0  = (const float*)d_in[4];   // [2,3072,1024]
  const float* bih0  = (const float*)d_in[5];   // [2,3072]
  const float* bhh0  = (const float*)d_in[6];   // [2,3072]
  const float* wih   = (const float*)d_in[7];   // [2,2,3072,2048]
  const float* whh   = (const float*)d_in[8];   // [2,2,3072,1024]
  const float* bih   = (const float*)d_in[9];   // [2,2,3072]
  const float* bhh   = (const float*)d_in[10];  // [2,2,3072]
  const float* fcw   = (const float*)d_in[11];  // [60,2048]
  const float* fcb   = (const float*)d_in[12];  // [60]
  float* out = (float*)d_out;
  char* ws = (char*)d_ws;

  // ---- workspace carve-up (all chunk sizes are multiples of 256B) ----
  size_t off = 0;
  float* meanp = (float*)(ws + off);
  float* rstdp = (float*)(ws + off + 1024);
  off += 4096;
  __bf16* xn    = (__bf16*)(ws + off); off += (size_t)T * NB * KP0 * 2;       // 3.07 MB
  __bf16* wih0b = (__bf16*)(ws + off); off += (size_t)2 * G3 * KP0 * 2;       // 1.97 MB
  __bf16* wihb  = (__bf16*)(ws + off); off += (size_t)2 * 2 * G3 * K2H * 2;   // 50.3 MB
  __bf16* whhb  = (__bf16*)(ws + off); off += (size_t)3 * 2 * G3 * H * 2;     // 37.7 MB
  __bf16* gibuf = (__bf16*)(ws + off); off += (size_t)2 * ROWS * G3 * 2;      // 118 MB
  __bf16* seqA  = (__bf16*)(ws + off); off += (size_t)T * NB * K2H * 2;       // 39.3 MB
  __bf16* seqB  = (__bf16*)(ws + off); off += (size_t)T * NB * K2H * 2;       // 39.3 MB
  float*  hAf   = (float*)(ws + off);  off += (size_t)2 * NB * H * 4;
  float*  hBf   = (float*)(ws + off);  off += (size_t)2 * NB * H * 4;
  __bf16* hAb   = (__bf16*)(ws + off); off += (size_t)2 * NB * H * 2;
  __bf16* hBb   = (__bf16*)(ws + off); off += (size_t)2 * NB * H * 2;

  // ---- BatchNorm + transpose into padded bf16 ----
  bn_stats_k<<<CIN, 256, 0, stream>>>(x, meanp, rstdp);
  bn_apply_k<<<(T * NB * KP0 + 255) / 256, 256, 0, stream>>>(x, meanp, rstdp, gamma, beta, xn);

  // ---- weight conversions (once per call; deterministic) ----
  cvt_pad_k<<<(2 * G3 * KP0 + 255) / 256, 256, 0, stream>>>(wih0, wih0b, 2 * G3, CIN, KP0);
  cvt_k<<<(2 * 2 * G3 * K2H + 255) / 256, 256, 0, stream>>>(wih, wihb, 2 * 2 * G3 * K2H);
  cvt_k<<<(2 * G3 * H + 255) / 256, 256, 0, stream>>>(whh0, whhb, 2 * G3 * H);
  cvt_k<<<(2 * 2 * G3 * H + 255) / 256, 256, 0, stream>>>(whh, whhb + (size_t)2 * G3 * H,
                                                          2 * 2 * G3 * H);

  const dim3 ggrid(G3 / 128, ROWS / 128, 2);
  const dim3 sgrid(H / 128, 1, 2);

  for (int l = 0; l < 3; l++) {
    const __bf16* Ain; int K; const __bf16* Wl; const float* bihl;
    __bf16* seqO;
    if (l == 0)      { Ain = xn;   K = KP0; Wl = wih0b;                      bihl = bih0;         seqO = seqA; }
    else if (l == 1) { Ain = seqA; K = K2H; Wl = wihb;                       bihl = bih;          seqO = seqB; }
    else             { Ain = seqB; K = K2H; Wl = wihb + (size_t)2 * G3 * K2H; bihl = bih + 2 * G3; seqO = seqA; }

    // gi = A @ W^T + b_ih  (bulk of the FLOPs, fully parallel WMMA GEMM)
    gi_gemm_k<<<ggrid, 256, 0, stream>>>(Ain, Wl, bihl, gibuf, K);

    // zero h0 (both f32 master and bf16 matmul copy)
    hipMemsetAsync(hAf, 0, (size_t)2 * NB * H * 4, stream);
    hipMemsetAsync(hAb, 0, (size_t)2 * NB * H * 2, stream);

    const __bf16* whhl = whhb + (size_t)l * 2 * G3 * H;
    const float* bhhl = (l == 0) ? bhh0 : (bhh + (size_t)(l - 1) * 2 * G3);

    for (int t = 0; t < T; t++) {
      bool even = (t & 1) == 0;   // double-buffer h across the sequential chain
      gru_step_k<<<sgrid, 256, 0, stream>>>(
          even ? hAb : hBb, even ? hAf : hBf,
          even ? hBb : hAb, even ? hBf : hAf,
          whhl, bhhl, gibuf, seqO, t);
    }
  }

  // last timestep of layer-2 output lives in seqA
  fc_k<<<(NB * NC + 255) / 256, 256, 0, stream>>>(
      seqA + (size_t)(T - 1) * NB * K2H, fcw, fcb, out);
}
